// SparseExpertLayer_36790689858310
// MI455X (gfx1250) — matrix operand order repaired and checked by reading, not compiled, not run on hardware
//
#include <hip/hip_runtime.h>
#include <math.h>

#define NTOK  4096          // B*T
#define CDIM  1024
#define EXP   8
#define HDIM  4096
#define TILEM 128
#define KSTEP 64
#define LDSK  72            // KSTEP + 8 pad (keeps 16B alignment: 72*2=144 bytes/row)
#define MAXR  (2*NTOK + EXP*TILEM)   // 9216 padded routed rows

typedef __attribute__((ext_vector_type(16))) __bf16 v16bf;
typedef __attribute__((ext_vector_type(8)))  __bf16 v8bf;
typedef __attribute__((ext_vector_type(4)))  __bf16 v4bf;
typedef __attribute__((ext_vector_type(8)))  float  v8f;

// ---------------- init: counters + row->token map ----------------
__global__ void k_init(int* cnt, int* cnt2, int* tokOf) {
  int i = blockIdx.x * blockDim.x + threadIdx.x;
  if (i < MAXR) tokOf[i] = -1;
  if (i < EXP) { cnt[i] = 0; cnt2[i] = 0; }
}

// ---------------- gating: one wave per token ----------------
__global__ __launch_bounds__(256)
void k_gate(const float* __restrict__ x, const float* __restrict__ Wg,
            const float* __restrict__ bg, int* __restrict__ idxTok,
            float* __restrict__ wgt, int* cnt) {
  int wave = threadIdx.x >> 5;
  int lane = threadIdx.x & 31;
  int t = blockIdx.x * 8 + wave;
  if (t >= NTOK) return;
  float acc[EXP];
#pragma unroll
  for (int e = 0; e < EXP; ++e) acc[e] = 0.f;
  const float* xr = x + (size_t)t * CDIM;
  for (int c = lane; c < CDIM; c += 32) {
    float xv = xr[c];
    const float* wr = Wg + (size_t)c * EXP;
#pragma unroll
    for (int e = 0; e < EXP; ++e) acc[e] += xv * wr[e];
  }
#pragma unroll
  for (int e = 0; e < EXP; ++e) {
#pragma unroll
    for (int m = 16; m >= 1; m >>= 1) acc[e] += __shfl_xor(acc[e], m, 32);
  }
  if (lane == 0) {
    float l[EXP];
#pragma unroll
    for (int e = 0; e < EXP; ++e) l[e] = acc[e] + bg[e];
    int e0 = 0;
#pragma unroll
    for (int e = 1; e < EXP; ++e) if (l[e] > l[e0]) e0 = e;
    int e1 = (e0 == 0) ? 1 : 0;
#pragma unroll
    for (int e = 0; e < EXP; ++e) if (e != e0 && l[e] > l[e1]) e1 = e;
    float w0 = 1.f / (1.f + __expf(l[e1] - l[e0]));   // normalized top-2 softmax
    idxTok[t * 2 + 0] = e0; idxTok[t * 2 + 1] = e1;
    wgt[t * 2 + 0] = w0;    wgt[t * 2 + 1] = 1.f - w0;
    atomicAdd(&cnt[e0], 1);
    atomicAdd(&cnt[e1], 1);
  }
}

// ---------------- expert base offsets (tile-aligned) ----------------
__global__ void k_bases(const int* cnt, int* basep) {
  if (threadIdx.x == 0 && blockIdx.x == 0) {
    int b = 0;
    for (int e = 0; e < EXP; ++e) {
      basep[e] = b;
      b += (cnt[e] + TILEM - 1) & ~(TILEM - 1);
    }
    basep[EXP] = b;
  }
}

// ---------------- scatter tokens to compacted expert rows ----------------
__global__ void k_scatter(const int* __restrict__ idxTok, const int* __restrict__ basep,
                          int* cnt2, int* __restrict__ tokOf, int* __restrict__ rowOfTok) {
  int t = blockIdx.x * blockDim.x + threadIdx.x;
  if (t >= NTOK) return;
#pragma unroll
  for (int k = 0; k < 2; ++k) {
    int e = idxTok[t * 2 + k];
    int pos = atomicAdd(&cnt2[e], 1);
    int r = basep[e] + pos;
    tokOf[r] = t;
    rowOfTok[t * 2 + k] = r;
  }
}

// Build v16bf A fragment from two aligned 16B LDS chunks (CDNA5 16-bit A 16x32 layout)
static __device__ __forceinline__ v16bf load_a_frag(const __bf16* rowp, int kc, int half) {
  v8bf lo = *(const v8bf*)(rowp + kc * 32 + half * 8);
  v8bf hi = *(const v8bf*)(rowp + kc * 32 + 16 + half * 8);
  return __builtin_shufflevector(lo, hi, 0, 1, 2, 3, 4, 5, 6, 7,
                                 8, 9, 10, 11, 12, 13, 14, 15);
}
// B fragment: lane's column strip is contiguous in transposed LDS tile
static __device__ __forceinline__ v16bf load_b_frag(const __bf16* colp, int kc, int half) {
  v8bf lo = *(const v8bf*)(colp + kc * 32 + half * 8);
  v8bf hi = *(const v8bf*)(colp + kc * 32 + 16 + half * 8);
  return __builtin_shufflevector(lo, hi, 0, 1, 2, 3, 4, 5, 6, 7,
                                 8, 9, 10, 11, 12, 13, 14, 15);
}

// ---------------- GEMM1: h = gelu(x_gathered @ W1[e] + b1[e]) ----------------
__global__ __launch_bounds__(256)
void k_gemm1(const float* __restrict__ x, const float* __restrict__ W1,
             const float* __restrict__ b1, const int* __restrict__ basep,
             const int* __restrict__ tokOf, __bf16* __restrict__ h) {
  __shared__ __bf16 As[TILEM][LDSK];     // 128 x 64 bf16, row-major (M x K)
  __shared__ __bf16 Bst[128][LDSK];      // 128 x 64 bf16, transposed (N x K)
  int m0 = blockIdx.y * TILEM;
  int n0 = blockIdx.x * 128;
  if (m0 >= basep[EXP]) return;
  int e = 0;
#pragma unroll
  for (int i = 1; i < EXP; ++i) if (m0 >= basep[i]) e = i;
  const float* W = W1 + (size_t)e * CDIM * HDIM;

  int tid = threadIdx.x;
  int lane = tid & 31, wave = tid >> 5;
  int waveM = wave >> 2, waveN = wave & 3;   // 2 x 4 wave grid, 64x32 per wave
  int half = lane >> 4, nrow = lane & 15;

  int cB = tid & 127;            // B staging: this thread's output column
  int rB = (tid >> 7) * 32;      // and K-strip base (0 or 32)

  int tokCache[8];
#pragma unroll
  for (int it = 0; it < 8; ++it)
    tokCache[it] = tokOf[m0 + (tid >> 4) + it * 16];

  v8f acc[4][2] = {};
  for (int kk = 0; kk < CDIM; kk += KSTEP) {
    // stage A: gathered token rows, float4 loads, branchless zero for pad rows
#pragma unroll
    for (int it = 0; it < 8; ++it) {
      int idx = tid + it * 256;            // 2048 float4 units = 128 x 16
      int r = idx >> 4, c4 = idx & 15;
      int tok = tokCache[it];
      int tsafe = tok < 0 ? 0 : tok;
      float4 f = *(const float4*)(x + (size_t)tsafe * CDIM + kk + c4 * 4);
      float s = tok < 0 ? 0.f : 1.f;
      v4bf b4 = { (__bf16)(f.x * s), (__bf16)(f.y * s),
                  (__bf16)(f.z * s), (__bf16)(f.w * s) };
      *(v4bf*)&As[r][c4 * 4] = b4;
    }
    // stage B transposed: coalesced global reads, vector LDS writes
    {
      __bf16 tmp[32];
#pragma unroll
      for (int i = 0; i < 32; ++i)
        tmp[i] = (__bf16)W[(size_t)(kk + rB + i) * HDIM + n0 + cB];
#pragma unroll
      for (int j = 0; j < 4; ++j)
        *(v8bf*)&Bst[cB][rB + j * 8] = *(v8bf*)&tmp[j * 8];
    }
    if (kk + KSTEP < CDIM)
      __builtin_prefetch(&W[(size_t)(kk + KSTEP + rB) * HDIM + n0 + cB], 0, 1);
    __syncthreads();

#pragma unroll
    for (int kc = 0; kc < 2; ++kc) {
      v16bf b[2];
#pragma unroll
      for (int in = 0; in < 2; ++in)
        b[in] = load_b_frag(&Bst[waveN * 32 + in * 16 + nrow][0], kc, half);
#pragma unroll
      for (int im = 0; im < 4; ++im) {
        v16bf a = load_a_frag(&As[waveM * 64 + im * 16 + nrow][0], kc, half);
#pragma unroll
        for (int in = 0; in < 2; ++in)
          acc[im][in] = __builtin_amdgcn_wmma_f32_16x16x32_bf16(
              false, a, false, b[in], (short)0, acc[im][in], false, false);
      }
    }
    __syncthreads();
  }

  const float* bias = b1 + (size_t)e * HDIM;
#pragma unroll
  for (int im = 0; im < 4; ++im)
#pragma unroll
    for (int in = 0; in < 2; ++in)
#pragma unroll
      for (int j = 0; j < 8; ++j) {
        int m = waveM * 64 + im * 16 + half * 8 + j;  // C/D: VGPR j -> M=j / j+8
        int n = waveN * 32 + in * 16 + nrow;
        float v = acc[im][in][j] + bias[n0 + n];
        v = 0.5f * v * (1.f + erff(v * 0.70710678118f));   // exact gelu
        h[(size_t)(m0 + m) * HDIM + n0 + n] = (__bf16)v;
      }
}

// ---------------- GEMM2: y = h @ W2[e] + b2[e] ----------------
__global__ __launch_bounds__(256)
void k_gemm2(const __bf16* __restrict__ h, const float* __restrict__ W2,
             const float* __restrict__ b2, const int* __restrict__ basep,
             float* __restrict__ y) {
  __shared__ __bf16 As[TILEM][LDSK];
  __shared__ __bf16 Bst[128][LDSK];
  int m0 = blockIdx.y * TILEM;
  int n0 = blockIdx.x * 128;
  if (m0 >= basep[EXP]) return;
  int e = 0;
#pragma unroll
  for (int i = 1; i < EXP; ++i) if (m0 >= basep[i]) e = i;
  const float* W = W2 + (size_t)e * HDIM * CDIM;

  int tid = threadIdx.x;
  int lane = tid & 31, wave = tid >> 5;
  int waveM = wave >> 2, waveN = wave & 3;
  int half = lane >> 4, nrow = lane & 15;

  int cB = tid & 127;
  int rB = (tid >> 7) * 32;

  v8f acc[4][2] = {};
  for (int kk = 0; kk < HDIM; kk += KSTEP) {
    // stage A: bf16 rows of h, 16B vector copies
#pragma unroll
    for (int it = 0; it < 4; ++it) {
      int idx = tid + it * 256;            // 1024 v8bf units = 128 x 8
      int r = idx >> 3, c8 = idx & 7;
      *(v8bf*)&As[r][c8 * 8] =
          *(const v8bf*)(h + (size_t)(m0 + r) * HDIM + kk + c8 * 8);
    }
    // stage B transposed
    {
      __bf16 tmp[32];
#pragma unroll
      for (int i = 0; i < 32; ++i)
        tmp[i] = (__bf16)W[(size_t)(kk + rB + i) * CDIM + n0 + cB];
#pragma unroll
      for (int j = 0; j < 4; ++j)
        *(v8bf*)&Bst[cB][rB + j * 8] = *(v8bf*)&tmp[j * 8];
    }
    if (kk + KSTEP < HDIM)
      __builtin_prefetch(&W[(size_t)(kk + KSTEP + rB) * CDIM + n0 + cB], 0, 1);
    __syncthreads();

#pragma unroll
    for (int kc = 0; kc < 2; ++kc) {
      v16bf b[2];
#pragma unroll
      for (int in = 0; in < 2; ++in)
        b[in] = load_b_frag(&Bst[waveN * 32 + in * 16 + nrow][0], kc, half);
#pragma unroll
      for (int im = 0; im < 4; ++im) {
        v16bf a = load_a_frag(&As[waveM * 64 + im * 16 + nrow][0], kc, half);
#pragma unroll
        for (int in = 0; in < 2; ++in)
          acc[im][in] = __builtin_amdgcn_wmma_f32_16x16x32_bf16(
              false, a, false, b[in], (short)0, acc[im][in], false, false);
      }
    }
    __syncthreads();
  }

  const float* bias = b2 + (size_t)e * CDIM;
#pragma unroll
  for (int im = 0; im < 4; ++im)
#pragma unroll
    for (int in = 0; in < 2; ++in)
#pragma unroll
      for (int j = 0; j < 8; ++j) {
        int m = waveM * 64 + im * 16 + half * 8 + j;
        int n = waveN * 32 + in * 16 + nrow;
        y[(size_t)(m0 + m) * CDIM + n0 + n] = acc[im][in][j] + bias[n0 + n];
      }
}

// ---------------- combine: out[t] = w0*y[r0] + w1*y[r1] ----------------
__global__ void k_combine(const float* __restrict__ y, const int* __restrict__ rowOfTok,
                          const float* __restrict__ wgt, float* __restrict__ out) {
  int i = blockIdx.x * blockDim.x + threadIdx.x;
  if (i >= NTOK * CDIM) return;
  int t = i >> 10;       // / CDIM
  int c = i & (CDIM - 1);
  int r0 = rowOfTok[t * 2], r1 = rowOfTok[t * 2 + 1];
  out[i] = wgt[t * 2] * y[(size_t)r0 * CDIM + c] +
           wgt[t * 2 + 1] * y[(size_t)r1 * CDIM + c];
}

extern "C" void kernel_launch(void* const* d_in, const int* in_sizes, int n_in,
                              void* d_out, int out_size, void* d_ws, size_t ws_size,
                              hipStream_t stream) {
  (void)in_sizes; (void)n_in; (void)out_size; (void)ws_size;
  const float* x  = (const float*)d_in[0];
  const float* Wg = (const float*)d_in[1];
  const float* bg = (const float*)d_in[2];
  const float* W1 = (const float*)d_in[3];
  const float* b1 = (const float*)d_in[4];
  const float* W2 = (const float*)d_in[5];
  const float* b2 = (const float*)d_in[6];
  float* out = (float*)d_out;

  char* ws = (char*)d_ws;
  size_t off = 0;
  auto carve = [&](size_t bytes) -> void* {
    void* p = ws + off;
    off += (bytes + 255) & ~(size_t)255;
    return p;
  };
  int*    cnt      = (int*)carve(EXP * sizeof(int));
  int*    cnt2     = (int*)carve(EXP * sizeof(int));
  int*    basep    = (int*)carve((EXP + 1) * sizeof(int));
  int*    idxTok   = (int*)carve((size_t)NTOK * 2 * sizeof(int));
  float*  wgt      = (float*)carve((size_t)NTOK * 2 * sizeof(float));
  int*    rowOfTok = (int*)carve((size_t)NTOK * 2 * sizeof(int));
  int*    tokOf    = (int*)carve((size_t)MAXR * sizeof(int));
  __bf16* h        = (__bf16*)carve((size_t)MAXR * HDIM * sizeof(__bf16));   // ~75.5 MB
  float*  y        = (float*)carve((size_t)MAXR * CDIM * sizeof(float));     // ~37.7 MB

  k_init<<<(MAXR + 255) / 256, 256, 0, stream>>>(cnt, cnt2, tokOf);
  k_gate<<<NTOK / 8, 256, 0, stream>>>(x, Wg, bg, idxTok, wgt, cnt);
  k_bases<<<1, 1, 0, stream>>>(cnt, basep);
  k_scatter<<<NTOK / 256, 256, 0, stream>>>(idxTok, basep, cnt2, tokOf, rowOfTok);
  k_gemm1<<<dim3(HDIM / 128, MAXR / TILEM), 256, 0, stream>>>(x, W1, b1, basep, tokOf, h);
  k_gemm2<<<dim3(CDIM / 128, MAXR / TILEM), 256, 0, stream>>>(h, W2, b2, basep, y);
  k_combine<<<(NTOK * CDIM) / 256, 256, 0, stream>>>(y, rowOfTok, wgt, out);
}
